// VoiceToLatent_20134806684401
// MI455X (gfx1250) — compile-verified, matrix operand order checked by reading, not run on hardware
//
#include <hip/hip_runtime.h>

typedef __attribute__((ext_vector_type(16))) __bf16 bfx16;
typedef __attribute__((ext_vector_type(8)))  float  fx8;
typedef __attribute__((ext_vector_type(4)))  unsigned int v4u;
typedef __attribute__((ext_vector_type(8)))  int v8i;
typedef __attribute__((ext_vector_type(4)))  int v4i;
typedef unsigned int u32;
typedef unsigned long long u64;

// ---------------------------------------------------------------------------
// WMMA helpers (CDNA5 wave32, V_WMMA_F32_16X16X32_BF16)
// ---------------------------------------------------------------------------
__device__ __forceinline__ fx8 wmma_bf16(bfx16 a, bfx16 b, fx8 c) {
  // (neg_a, A, neg_b, B, c_mod, C, reuse_a, reuse_b)
  return __builtin_amdgcn_wmma_f32_16x16x32_bf16(false, a, false, b, (short)0, c, false, false);
}

// K position of the even element of VGPR-slot v for this lane, per the
// documented 16-bit 16x32 A layout.
__device__ __forceinline__ int kpos_base(int lane, int v) {
  return (v >> 2) * 16 + (lane >> 4) * 8 + (v & 3) * 2;
}

// Issue a 2D TENSOR_LOAD_TO_LDS: tile rows x cols (bf16), global row stride
// `strideElems`, hardware zero-fill for rows >= validRows (OOB semantics).
__device__ __forceinline__ void tdm_load_2d(u32 ldsAddr, const __bf16* gptr,
                                            int cols, int rows, int validRows,
                                            int strideElems)
{
  u64 ga = (u64)(size_t)gptr;
  v4u g0;
  g0.x = 1u;                                   // count=1, user mode, no gather
  g0.y = ldsAddr;                              // LDS byte address
  g0.z = (u32)ga;                              // global_addr[31:0]
  g0.w = (u32)((ga >> 32) & 0x1FFFFFFu) | (2u << 30);  // addr[56:32] | type=2
  v8i g1;
  g1[0] = (int)(1u << 16);                     // data_size=1 (2 bytes)
  g1[1] = (int)(((u32)cols & 0xFFFFu) << 16);  // tensor_dim0[15:0]
  g1[2] = (int)((((u32)cols >> 16) & 0xFFFFu) | (((u32)validRows & 0xFFFFu) << 16));
  g1[3] = (int)((((u32)validRows >> 16) & 0xFFFFu) | (((u32)cols & 0xFFFFu) << 16)); // tile_dim0
  g1[4] = (int)((u32)rows & 0xFFFFu);          // tile_dim1 (tile_dim2=0)
  g1[5] = (int)(u32)strideElems;               // tensor_dim0_stride[31:0]
  g1[6] = 0;                                   // stride0[47:32] | stride1[15:0]
  g1[7] = 0;
  v4i gz = {0, 0, 0, 0};
#if __clang_major__ >= 23
  v8i gz8 = {0, 0, 0, 0, 0, 0, 0, 0};
  __builtin_amdgcn_tensor_load_to_lds(g0, g1, gz, gz, gz8, 0);
#else
  __builtin_amdgcn_tensor_load_to_lds(g0, g1, gz, gz, 0);
#endif
}

// ---------------------------------------------------------------------------
// Weight packing into per-lane WMMA B-fragment order (bf16)
// ---------------------------------------------------------------------------
__global__ void pack_conv_w(const float* __restrict__ w, __bf16* __restrict__ wp,
                            int Cin, int CI_BLK, int CPB, int chunksTotal, int total)
{
  int idx = blockIdx.x * blockDim.x + threadIdx.x;
  if (idx >= total) return;
  int e      = idx & 15;
  int lane   = (idx >> 4) & 31;
  int cG     = (idx >> 9) % chunksTotal;
  int coTile = idx / (chunksTotal << 9);
  int blk = cG / CPB, cl = cG - blk * CPB;
  int kk  = cl * 32 + kpos_base(lane, e >> 1) + (e & 1);
  int k   = kk / CI_BLK;
  int ci  = blk * CI_BLK + (kk % CI_BLK);
  int co  = coTile * 16 + (lane & 15);
  float val = 0.f;
  if (k < 5) val = w[((size_t)co * Cin + ci) * 5 + k];
  wp[idx] = (__bf16)val;
}

__global__ void pack_dense(const float* __restrict__ w, __bf16* __restrict__ wp,
                           int Krows, int chunks, int secTiles, int secLen, int total)
{
  int idx = blockIdx.x * blockDim.x + threadIdx.x;
  if (idx >= total) return;
  int e     = idx & 15;
  int lane  = (idx >> 4) & 31;
  int c     = (idx >> 9) % chunks;
  int nTile = idx / (chunks << 9);
  int kk  = c * 32 + kpos_base(lane, e >> 1) + (e & 1);
  int sec = nTile / secTiles;
  int jl  = (nTile - sec * secTiles) * 16 + (lane & 15);
  float val = 0.f;
  if (kk < Krows && jl < secLen)
    val = w[((size_t)(sec * secLen + jl)) * Krows + kk];
  wp[idx] = (__bf16)val;
}

// ---------------------------------------------------------------------------
// conv0: Cin=1, Cout=16 — trivial FMA kernel, emits bf16 activations [b][t][16]
// ---------------------------------------------------------------------------
__global__ void conv0_kernel(const float* __restrict__ in, const float* __restrict__ w,
                             const float* __restrict__ bias, __bf16* __restrict__ out,
                             int Tin, int Tout)
{
  int idx = blockIdx.x * blockDim.x + threadIdx.x;
  if (idx >= 32 * Tout) return;
  int b = idx / Tout, t = idx - b * Tout;
  const float* ip = in + (size_t)b * Tin + 2 * t;
  float x0 = ip[0], x1 = ip[1], x2 = ip[2], x3 = ip[3], x4 = ip[4];
  __bf16* op = out + ((size_t)b * Tout + t) * 16;
#pragma unroll
  for (int co = 0; co < 16; ++co) {
    const float* wc = w + co * 5;
    float s = bias[co] + x0 * wc[0] + x1 * wc[1] + x2 * wc[2] + x3 * wc[3] + x4 * wc[4];
    op[co] = (__bf16)(s > 0.f ? s : 0.f);
  }
}

// ---------------------------------------------------------------------------
// Implicit-GEMM conv (layers 1..6), bf16 WMMA, fp32 accumulate.
// Input slab (68 samples x CI_BLK channels) double-buffered in LDS and moved
// by the Tensor Data Mover (one TENSOR_LOAD_TO_LDS per ci-block, TENSORcnt),
// overlapping the DMA of block k+1 with the WMMAs of block k. OOB time rows
// are zero-filled by the TDM's tensor-dim clamp.
// ---------------------------------------------------------------------------
template <int CI_BLK>
__global__ void __launch_bounds__(256)
conv_wmma_kernel(const __bf16* __restrict__ in, const __bf16* __restrict__ wp,
                 const float* __restrict__ bias, __bf16* __restrict__ out,
                 int Cin, int Cout, int Tin, int Tout)
{
  constexpr int CPB   = (5 * CI_BLK + 31) / 32;   // K-chunks per ci-block
  constexpr int SLABW = (68 * CI_BLK) / 2;        // u32 words per slab
  __shared__ u32 slab_u[2][SLABW];

  const int b      = blockIdx.z;
  const int t0     = blockIdx.x * 32;
  const int coBase = blockIdx.y * 64;
  const int tid  = threadIdx.x;
  const int lane = tid & 31;
  const int wave = tid >> 5;
  const int mSub = wave & 1;
  const int nSub = wave >> 1;
  const int nBlk = Cin / CI_BLK;
  const int chunksTotal = nBlk * CPB;
  const int coTile = (coBase >> 4) + nSub;
  const bfx16* wpv = (const bfx16*)wp;

  const int m    = lane & 15;
  const int half = lane >> 4;
  const int mloc = mSub * 16 + m;

  int validRows = Tin - t0 * 2;
  if (validRows > 68) validRows = 68;

  auto stage = [&](int blk, u32* buf) {
    if (wave == 0) {
      const __bf16* g = in + ((size_t)b * Tin + t0 * 2) * Cin + blk * CI_BLK;
      tdm_load_2d((u32)(size_t)buf, g, CI_BLK, 68, validRows, Cin);
    }
  };

  fx8 acc = {};
  stage(0, slab_u[0]);
  if (wave == 0) __builtin_amdgcn_s_wait_tensorcnt(0);
  __syncthreads();

  for (int blk = 0; blk < nBlk; ++blk) {
    const u32* cur = slab_u[blk & 1];
    if (blk + 1 < nBlk) stage(blk + 1, slab_u[(blk + 1) & 1]);

    for (int c = 0; c < CPB; ++c) {
      bfx16 a;
      u32* au = (u32*)&a;
#pragma unroll
      for (int v = 0; v < 8; ++v) {
        int kk  = c * 32 + kpos_base(lane, v);
        int k   = kk / CI_BLK;
        int cil = kk & (CI_BLK - 1);
        u32 val = 0u;
        if (k < 5) val = cur[((2 * mloc + k) * CI_BLK + cil) >> 1];
        au[v] = val;
      }
      bfx16 bf = wpv[((size_t)coTile * chunksTotal + blk * CPB + c) * 32 + lane];
      acc = wmma_bf16(a, bf, acc);
    }
    if (wave == 0) __builtin_amdgcn_s_wait_tensorcnt(0);
    __syncthreads();
  }

  // epilogue: C layout — VGPR j: lanes0-15 -> M=j, lanes16-31 -> M=j+8
  const int n  = lane & 15;
  const int co = coBase + nSub * 16 + n;
  const float bv = bias[co];
#pragma unroll
  for (int j = 0; j < 8; ++j) {
    int t = t0 + mSub * 16 + j + 8 * half;
    if (t < Tout) {
      float v = acc[j] + bv;
      out[((size_t)b * Tout + t) * Cout + co] = (__bf16)(v > 0.f ? v : 0.f);
    }
  }
}

// ---------------------------------------------------------------------------
// conv_out transpose: [b][t][512] bf16 -> d_out [t][b][512] fp32
// ---------------------------------------------------------------------------
__global__ void conv_out_transpose(const __bf16* __restrict__ act, float* __restrict__ out, int T)
{
  int idx = blockIdx.x * blockDim.x + threadIdx.x;
  if (idx >= T * 32 * 512) return;
  int c = idx & 511;
  int b = (idx >> 9) & 31;
  int t = idx >> 14;
  out[idx] = (float)act[(((size_t)b * T) + t) * 512 + c];
}

// ---------------------------------------------------------------------------
// Persistent GRU kernel: one workgroup (32 waves), all 509 steps.
// ---------------------------------------------------------------------------
#define GRU_CELL(XIN_U, HB_U, HF, PIH, PHH, BIH, BHH, HAS_XNEXT, XNEXT, STPTR, HAS_OUT, OUTPTR)    \
  {                                                                                                \
    for (int task = wave; task < 66; task += 32) {                                                 \
      int mT = task / 33, nT = task - mT * 33;                                                     \
      fx8 acc = {};                                                                                \
      int arow = (mT * 16 + m) * 192;                                                              \
      for (int c = 0; c < 6; ++c) {                                                                \
        bfx16 a; u32* au = (u32*)&a;                                                               \
        _Pragma("unroll")                                                                          \
        for (int v = 0; v < 8; ++v)                                                                \
          au[v] = HB_U[(arow + c * 32 + kpos_base(lane, v)) >> 1];                                 \
        acc = wmma_bf16(a, PHH[(nT * 6 + c) * 32 + lane], acc);                                    \
      }                                                                                            \
      int jp = nT * 16 + n;                                                                        \
      int sec = jp / 176, jl = jp - sec * 176;                                                     \
      float bb = (jl < 175) ? BHH[sec * 175 + jl] : 0.f;                                           \
      _Pragma("unroll")                                                                            \
      for (int j = 0; j < 8; ++j) {                                                                \
        int row = mT * 16 + j + 8 * half;                                                          \
        ghbuf[row * 528 + jp] = acc[j] + bb;                                                       \
      }                                                                                            \
    }                                                                                              \
    __syncthreads();                                                                               \
    for (int task = wave; task < 22; task += 32) {                                                 \
      int mT = task / 11, nT = task - mT * 11;                                                     \
      fx8 aR = {}, aZ = {}, aN = {};                                                               \
      int arow = (mT * 16 + m) * 192;                                                              \
      for (int c = 0; c < 6; ++c) {                                                                \
        bfx16 a; u32* au = (u32*)&a;                                                               \
        _Pragma("unroll")                                                                          \
        for (int v = 0; v < 8; ++v)                                                                \
          au[v] = XIN_U[(arow + c * 32 + kpos_base(lane, v)) >> 1];                                \
        aR = wmma_bf16(a, PIH[((0 * 11 + nT) * 6 + c) * 32 + lane], aR);                           \
        aZ = wmma_bf16(a, PIH[((1 * 11 + nT) * 6 + c) * 32 + lane], aZ);                           \
        aN = wmma_bf16(a, PIH[((2 * 11 + nT) * 6 + c) * 32 + lane], aN);                           \
      }                                                                                            \
      int jl = nT * 16 + n;                                                                        \
      if (jl < 175) {                                                                              \
        float br = BIH[jl], bz = BIH[175 + jl], bn = BIH[350 + jl];                                \
        _Pragma("unroll")                                                                          \
        for (int j = 0; j < 8; ++j) {                                                              \
          int row = mT * 16 + j + 8 * half;                                                        \
          float ghr = ghbuf[row * 528 + jl];                                                       \
          float ghz = ghbuf[row * 528 + 176 + jl];                                                 \
          float ghn = ghbuf[row * 528 + 352 + jl];                                                 \
          float r  = 1.f / (1.f + __expf(-(aR[j] + br + ghr)));                                    \
          float z  = 1.f / (1.f + __expf(-(aZ[j] + bz + ghz)));                                    \
          float nn = tanhf(aN[j] + bn + r * ghn);                                                  \
          float hp = HF[row * 176 + jl];                                                           \
          float hv = (1.f - z) * nn + z * hp;                                                      \
          HF[row * 176 + jl] = hv;                                                                 \
          if (HAS_XNEXT) XNEXT[row * 192 + jl] = (__bf16)hv;                                       \
          (STPTR)[row * 175 + jl] = hv;                                                            \
          if (HAS_OUT) (OUTPTR)[row * 175 + jl] = hv;                                              \
        }                                                                                          \
      }                                                                                            \
    }                                                                                              \
  }

__global__ void __launch_bounds__(1024)
gru_kernel(const __bf16* __restrict__ act6,
           const bfx16* __restrict__ lp,
           const bfx16* __restrict__ pih0, const bfx16* __restrict__ phh0,
           const bfx16* __restrict__ pih1, const bfx16* __restrict__ phh1,
           const float* __restrict__ lin_b,
           const float* __restrict__ bih0, const float* __restrict__ bhh0,
           const float* __restrict__ bih1, const float* __restrict__ bhh1,
           float* __restrict__ outputs, float* __restrict__ states, int T)
{
  __shared__ float ghbuf[32 * 528];     // also hosts the 32x512 bf16 x-slice
  __shared__ float h0f[32 * 176];
  __shared__ float h1f[32 * 176];
  __shared__ u32 xt_u[32 * 96];
  __shared__ u32 x1_u[32 * 96];
  __shared__ u32 h0b_u[32 * 96];
  __shared__ u32 h1b_u[32 * 96];
  __bf16* xt  = (__bf16*)xt_u;
  __bf16* x1b = (__bf16*)x1_u;
  __bf16* h0b = (__bf16*)h0b_u;
  __bf16* h1b = (__bf16*)h1b_u;
  u32* xslab = (u32*)ghbuf;             // 32x512 bf16 = 8192 u32 (ghbuf is idle here)

  const int tid = threadIdx.x, lane = tid & 31, wave = tid >> 5;
  for (int i = tid; i < 32 * 96; i += 1024) { xt_u[i] = 0; x1_u[i] = 0; h0b_u[i] = 0; h1b_u[i] = 0; }
  for (int i = tid; i < 32 * 176; i += 1024) { h0f[i] = 0.f; h1f[i] = 0.f; }
  __syncthreads();

  const int m = lane & 15, half = lane >> 4, n = lane & 15;
  for (int t = 0; t < T; ++t) {
    // 1) async-stage this step's x slice (32 rows x 512 bf16) into LDS while
    //    doing the h decay; ghbuf is free until cell0 phase A.
    for (int i = tid; i < 2048; i += 1024) {        // 2048 x 16B transfers
      int row = i >> 6, seg = i & 63;
      u32 lds = (u32)(size_t)(xslab + i * 4);
      const __bf16* g = act6 + ((size_t)row * T + t) * 512 + seg * 8;
      asm volatile("global_load_async_to_lds_b128 %0, %1, off"
                   :: "v"(lds), "v"(g) : "memory");
    }
    for (int i = tid; i < 32 * 176; i += 1024) {
      int b = i / 176, j = i - b * 176;
      float a0 = h0f[i] * 0.9f; h0f[i] = a0; h0b[b * 192 + j] = (__bf16)a0;
      float a1 = h1f[i] * 0.9f; h1f[i] = a1; h1b[b * 192 + j] = (__bf16)a1;
    }
    if (wave == 0 && t + 1 < T)
      __builtin_prefetch(act6 + ((size_t)lane * T + t + 1) * 512, 0, 1);
    asm volatile("s_wait_asynccnt 0x0" ::: "memory");
    __syncthreads();

    // 2) xt = relu(x @ lin_w^T + lin_b); A gathered from the LDS x slice
    for (int task = wave; task < 22; task += 32) {
      int mT = task / 11, nT = task - mT * 11;
      fx8 acc = {};
      int arow = (mT * 16 + m) * 512;
      for (int c = 0; c < 16; ++c) {
        bfx16 a; u32* au = (u32*)&a;
#pragma unroll
        for (int v = 0; v < 8; ++v)
          au[v] = xslab[(arow + c * 32 + kpos_base(lane, v)) >> 1];
        acc = wmma_bf16(a, lp[(nT * 16 + c) * 32 + lane], acc);
      }
      int jp = nT * 16 + n;
      if (jp < 175) {
        float bb = lin_b[jp];
#pragma unroll
        for (int j = 0; j < 8; ++j) {
          int row = mT * 16 + j + 8 * half;
          float v2 = acc[j] + bb;
          xt[row * 192 + jp] = (__bf16)(v2 > 0.f ? v2 : 0.f);
        }
      }
    }
    __syncthreads();

    // 3) GRU layer 0, then layer 1
    float* st0  = states + ((size_t)t * 2 + 0) * 32 * 175;
    float* st1  = states + ((size_t)t * 2 + 1) * 32 * 175;
    float* outp = outputs + (size_t)t * 32 * 175;
    GRU_CELL(xt_u, h0b_u, h0f, pih0, phh0, bih0, bhh0, 1, x1b, st0, 0, outp)
    __syncthreads();
    GRU_CELL(x1_u, h1b_u, h1f, pih1, phh1, bih1, bhh1, 0, x1b, st1, 1, outp)
    __syncthreads();
  }
}

// ---------------------------------------------------------------------------
// Host launcher
// ---------------------------------------------------------------------------
extern "C" void kernel_launch(void* const* d_in, const int* in_sizes, int n_in,
                              void* d_out, int out_size, void* d_ws, size_t ws_size,
                              hipStream_t stream)
{
  (void)in_sizes; (void)n_in; (void)out_size; (void)ws_size;
  const float* inputs = (const float*)d_in[0];
  const float* cw[7]; const float* cb[7];
  for (int l = 0; l < 7; ++l) { cw[l] = (const float*)d_in[1 + 2 * l]; cb[l] = (const float*)d_in[2 + 2 * l]; }
  const float* lin_w = (const float*)d_in[15];
  const float* lin_b = (const float*)d_in[16];
  const float* wih0 = (const float*)d_in[17];
  const float* whh0 = (const float*)d_in[18];
  const float* bih0 = (const float*)d_in[19];
  const float* bhh0 = (const float*)d_in[20];
  const float* wih1 = (const float*)d_in[21];
  const float* whh1 = (const float*)d_in[22];
  const float* bih1 = (const float*)d_in[23];
  const float* bhh1 = (const float*)d_in[24];
  float* out = (float*)d_out;

  static const int Tdim[8] = {65536, 32766, 16381, 8189, 4093, 2045, 1021, 509};
  static const int CH[8]   = {1, 16, 64, 128, 256, 512, 512, 512};

  size_t off = 0;
  auto carve = [&](size_t bytes) -> char* {
    char* p = (char*)d_ws + off; off += (bytes + 255) & ~(size_t)255; return p;
  };
  const size_t ACT_ELEMS = (size_t)32 * 16381 * 64;   // largest intermediate
  __bf16* actA = (__bf16*)carve(ACT_ELEMS * 2);
  __bf16* actB = (__bf16*)carve(ACT_ELEMS * 2);
  __bf16* cpack[7] = {};
  int cibArr[7], cpbArr[7], ctotArr[7];
  for (int l = 1; l <= 6; ++l) {
    int Cin = CH[l], Cout = CH[l + 1];
    int cib = Cin < 128 ? Cin : 128;
    int cpb = (5 * cib + 31) / 32;
    int ctot = (Cin / cib) * cpb;
    cibArr[l] = cib; cpbArr[l] = cpb; ctotArr[l] = ctot;
    cpack[l] = (__bf16*)carve((size_t)(Cout / 16) * ctot * 512 * 2);
  }
  __bf16* lpack = (__bf16*)carve((size_t)11 * 16 * 512 * 2);
  __bf16* pih0  = (__bf16*)carve((size_t)33 * 6 * 512 * 2);
  __bf16* phh0  = (__bf16*)carve((size_t)33 * 6 * 512 * 2);
  __bf16* pih1  = (__bf16*)carve((size_t)33 * 6 * 512 * 2);
  __bf16* phh1  = (__bf16*)carve((size_t)33 * 6 * 512 * 2);

  // --- weight packing ---
  for (int l = 1; l <= 6; ++l) {
    int total = (CH[l + 1] / 16) * ctotArr[l] * 512;
    pack_conv_w<<<(total + 255) / 256, 256, 0, stream>>>(
        cw[l], cpack[l], CH[l], cibArr[l], cpbArr[l], ctotArr[l], total);
  }
  {
    int total = 11 * 16 * 512;
    pack_dense<<<(total + 255) / 256, 256, 0, stream>>>(lin_w, lpack, 512, 16, 11, 175, total);
    total = 33 * 6 * 512;
    pack_dense<<<(total + 255) / 256, 256, 0, stream>>>(wih0, pih0, 175, 6, 11, 175, total);
    pack_dense<<<(total + 255) / 256, 256, 0, stream>>>(whh0, phh0, 175, 6, 11, 175, total);
    pack_dense<<<(total + 255) / 256, 256, 0, stream>>>(wih1, pih1, 175, 6, 11, 175, total);
    pack_dense<<<(total + 255) / 256, 256, 0, stream>>>(whh1, phh1, 175, 6, 11, 175, total);
  }

  // --- conv stack ---
  {
    int total = 32 * Tdim[1];
    conv0_kernel<<<(total + 255) / 256, 256, 0, stream>>>(inputs, cw[0], cb[0], actA, Tdim[0], Tdim[1]);
  }
  __bf16* bufs[2] = {actA, actB};
  for (int l = 1; l <= 6; ++l) {
    const __bf16* src = bufs[(l - 1) & 1];
    __bf16* dst = bufs[l & 1];
    dim3 grid((Tdim[l + 1] + 31) / 32, CH[l + 1] / 64, 32);
    if (cibArr[l] == 16)
      conv_wmma_kernel<16><<<grid, 256, 0, stream>>>(src, cpack[l], cb[l], dst,
                                                     CH[l], CH[l + 1], Tdim[l], Tdim[l + 1]);
    else if (cibArr[l] == 64)
      conv_wmma_kernel<64><<<grid, 256, 0, stream>>>(src, cpack[l], cb[l], dst,
                                                     CH[l], CH[l + 1], Tdim[l], Tdim[l + 1]);
    else
      conv_wmma_kernel<128><<<grid, 256, 0, stream>>>(src, cpack[l], cb[l], dst,
                                                      CH[l], CH[l + 1], Tdim[l], Tdim[l + 1]);
  }
  // conv6 output landed in actA ([b][t][512] bf16)

  // --- conv_out (fp32, [t][b][c]) ---
  {
    int total = Tdim[7] * 32 * 512;
    conv_out_transpose<<<(total + 255) / 256, 256, 0, stream>>>(actA, out, Tdim[7]);
  }

  // --- persistent GRU ---
  const size_t CONV_OUT = (size_t)Tdim[7] * 32 * 512;       // 8,339,456
  const size_t OUTPUTS  = (size_t)Tdim[7] * 32 * 175;       // 2,850,400
  gru_kernel<<<1, 1024, 0, stream>>>(actA, (const bfx16*)lpack,
                                     (const bfx16*)pih0, (const bfx16*)phh0,
                                     (const bfx16*)pih1, (const bfx16*)phh1,
                                     lin_b, bih0, bhh0, bih1, bhh1,
                                     out + CONV_OUT,                 // outputs
                                     out + CONV_OUT + OUTPUTS,       // states_hist
                                     Tdim[7]);
}
